// AutoregressiveMatrixChain_32899449487410
// MI455X (gfx1250) — matrix operand order repaired and checked by reading, not compiled, not run on hardware
//
#include <hip/hip_runtime.h>
#include <math.h>

// Problem constants (match reference)
#define B_  16
#define LL_ 512
#define LP_ 2048
#define H_  1024
#define C_  8192
#define S1_ 9      // S-1 slots
#define T_  4

// d_out layout (floats): summary[16][4][1024] | stop_logits[16][4] | stop_probs[16][4] | chain_len[16]
#define OFF_LOG  (B_*T_*H_)        // 65536
#define OFF_PRB  (OFF_LOG + B_*T_) // 65600
#define OFF_CHN  (OFF_PRB + B_*T_) // 65664

typedef __bf16 bf16_t;
typedef bf16_t v16bf __attribute__((ext_vector_type(16)));
typedef float  v8f   __attribute__((ext_vector_type(8)));

union ABreg { v16bf v; unsigned int u[8]; };

__device__ __forceinline__ unsigned short f2bf(float f){
  unsigned int x = __float_as_uint(f);
  unsigned int r = x + 0x7FFFu + ((x >> 16) & 1u);   // round-to-nearest-even
  return (unsigned short)(r >> 16);
}
__device__ __forceinline__ unsigned int bfpair(float lo, float hi){
  return (unsigned int)f2bf(lo) | ((unsigned int)f2bf(hi) << 16);
}
__device__ __forceinline__ float sigmoidf_(float x){ return 1.0f / (1.0f + __expf(-x)); }

// WMMA lane-layout helpers (CDNA5 ISA §7.12.2, bf16 16x16x32):
//  A (16x32): lane m (0-15)/(16-31 same M, K+16); VGPR j holds K pair 2j+half*16 as packed here (contiguous)
//  B (32x16): lane n = lane&15; VGPR j holds K pair half*16 + 2j as packed here, same as A (not the ISA table order)
//  C (16x16 f32): lane n = lane&15; VGPR j row m = j + half*8

// ---------------- means: out[b][off+h] = mean_l in[b][l][h] ----------------
__global__ void k_mean(const float* __restrict__ in, int L, float* __restrict__ out,
                       int outStride, int outOffset){
  int h = blockIdx.x * 256 + threadIdx.x;
  int b = blockIdx.y;
  const float* p = in + (size_t)b * L * H_ + h;
  float s = 0.f;
  for (int l = 0; l < L; ++l) s += p[(size_t)l * H_];
  out[(size_t)b * outStride + outOffset + h] = s / (float)L;
}

// ---------------- codebook: bf16 copy + row sum-of-squares ----------------
__global__ void k_cbprep(const float* __restrict__ cb, unsigned short* __restrict__ cb_bf,
                         float* __restrict__ cb_sq){
  __shared__ float red[256];
  int c = blockIdx.x, tid = threadIdx.x;
  const float* row = cb + (size_t)c * H_;
  float s = 0.f;
  for (int i = tid; i < H_; i += 256){ float v = row[i]; s += v*v; cb_bf[(size_t)c*H_ + i] = f2bf(v); }
  red[tid] = s; __syncthreads();
  for (int o = 128; o > 0; o >>= 1){ if (tid < o) red[tid] += red[tid+o]; __syncthreads(); }
  if (tid == 0) cb_sq[c] = red[0];
}

// ---------------- generic skinny GEMM: out[M,N] = act(A[M,K] @ W[K,N] + bias) ----------------
// grid (N/128, ceil(M/16)), block 256 (8 waves); LDS-staged bf16 weight tile.
__global__ __launch_bounds__(256) void k_gemm_small(
    const float* __restrict__ A, int M, int K,
    const float* __restrict__ W, int N,
    const float* __restrict__ bias,
    float* __restrict__ out, int act){
  __shared__ unsigned short ldsB[32 * 128];      // pair layout: (k>>1)*256 + n*2 + (k&1)
  unsigned int* ldsB32 = (unsigned int*)ldsB;
  int tid = threadIdx.x;
  int lane = tid & 31, wave = tid >> 5;
  int half = lane >> 4, mrow = lane & 15;
  int nblk = blockIdx.x * 128;
  int m0 = blockIdx.y * 16;
  int am = m0 + mrow; if (am > M-1) am = M-1;     // clamp (padded rows discarded)
  const float* Arow = A + (size_t)am * K;
  int nsub = wave * 16;
  v8f acc = {0.f,0.f,0.f,0.f,0.f,0.f,0.f,0.f};
  for (int k0 = 0; k0 < K; k0 += 32){
    #pragma unroll
    for (int i = 0; i < 16; ++i){
      int e = tid + i * 256;
      int kk = e >> 7, nn = e & 127;
      ldsB[(kk >> 1) * 256 + nn * 2 + (kk & 1)] = f2bf(W[(size_t)(k0 + kk) * N + nblk + nn]);
    }
    __syncthreads();
    ABreg a, b;
    #pragma unroll
    for (int j = 0; j < 8; ++j){
      int ka = k0 + 2*j + half * 16;
      a.u[j] = bfpair(Arow[ka], Arow[ka+1]);
      b.u[j] = ldsB32[(half * 8 + j) * 128 + nsub + mrow];
    }
    acc = __builtin_amdgcn_wmma_f32_16x16x32_bf16(false, a.v, false, b.v, (short)0, acc, false, false);
    __syncthreads();
  }
  int n = nblk + nsub + mrow;
  float bv = bias ? bias[n] : 0.f;               // hoisted: n is j-invariant
  #pragma unroll
  for (int j = 0; j < 8; ++j){
    int m = m0 + j + half * 8;
    if (m < M){
      float v = acc[j] + bv;
      if (act == 1) v = tanhf(v);
      out[(size_t)m * N + n] = v;
    }
  }
}

// ---------------- big projection: out_bf16[32768,1024] = A[32768,1024] @ W[1024,1024] ----------------
// grid (8, 512), block 256; 64x128 per block, LDS-staged A and B, prefetch next k-tile.
__global__ __launch_bounds__(256) void k_proj(
    const float* __restrict__ A, const float* __restrict__ W,
    unsigned short* __restrict__ out){
  __shared__ unsigned short ldsA[64 * 32];       // row-major bf16; u32 = (k,k+1) pair
  __shared__ unsigned short ldsB[32 * 128];      // pair layout
  unsigned int* ldsA32 = (unsigned int*)ldsA;
  unsigned int* ldsB32 = (unsigned int*)ldsB;
  int tid = threadIdx.x;
  int lane = tid & 31, wave = tid >> 5;
  int half = lane >> 4, mrow = lane & 15;
  int nblk = blockIdx.x * 128;
  int m0 = blockIdx.y * 64;
  int nsub = wave * 16;
  v8f acc[4];
  #pragma unroll
  for (int mt = 0; mt < 4; ++mt) acc[mt] = (v8f){0.f,0.f,0.f,0.f,0.f,0.f,0.f,0.f};
  for (int k0 = 0; k0 < H_; k0 += 32){
    // prefetch next k-tile (global_prefetch_b8)
    if (k0 + 32 < H_){
      if (tid < 64) __builtin_prefetch(A + (size_t)(m0 + tid) * H_ + k0 + 32, 0, 0);
      else if (tid < 96) __builtin_prefetch(W + (size_t)(k0 + 32 + (tid - 64)) * H_ + nblk, 0, 0);
    }
    #pragma unroll
    for (int i = 0; i < 8; ++i){
      int e = tid + i * 256;
      int mm = e >> 5, kk = e & 31;
      ldsA[mm * 32 + kk] = f2bf(A[(size_t)(m0 + mm) * H_ + k0 + kk]);
    }
    #pragma unroll
    for (int i = 0; i < 16; ++i){
      int e = tid + i * 256;
      int kk = e >> 7, nn = e & 127;
      ldsB[(kk >> 1) * 256 + nn * 2 + (kk & 1)] = f2bf(W[(size_t)(k0 + kk) * H_ + nblk + nn]);
    }
    __syncthreads();
    ABreg b;
    #pragma unroll
    for (int j = 0; j < 8; ++j)
      b.u[j] = ldsB32[(half * 8 + j) * 128 + nsub + mrow];
    #pragma unroll
    for (int mt = 0; mt < 4; ++mt){
      ABreg a;
      #pragma unroll
      for (int j = 0; j < 8; ++j){
        int kp = j + half * 8;     // pair index within 32-K tile
        a.u[j] = ldsA32[(mt * 16 + mrow) * 16 + kp];
      }
      acc[mt] = __builtin_amdgcn_wmma_f32_16x16x32_bf16(false, a.v, false, b.v, (short)0, acc[mt], false, false);
    }
    __syncthreads();
  }
  #pragma unroll
  for (int mt = 0; mt < 4; ++mt){
    #pragma unroll
    for (int j = 0; j < 8; ++j){
      int m = m0 + mt*16 + j + half * 8;
      int n = nblk + nsub + mrow;
      out[(size_t)m * H_ + n] = f2bf(acc[mt][j]);
    }
  }
}

// ---------------- scores: out[b][m][n] = scale * sum_k q[b][m][k]*rows[b][n][k] ----------------
// rows are bf16, row-contiguous (pk or codebook). K = H_. grid (Ncols/128, nbatch), block 256.
__global__ __launch_bounds__(256) void k_attn_scores(
    const float* __restrict__ q, int R, long qBS,
    const unsigned short* __restrict__ rows, long rBS,
    int Ncols, float* __restrict__ out, long outBS, float scale){
  int tid = threadIdx.x;
  int lane = tid & 31, wave = tid >> 5;
  int half = lane >> 4, mrow = lane & 15;
  int bidx = blockIdx.y;
  int r0 = blockIdx.x * 128 + wave * 16;
  int am = mrow; if (am > R-1) am = R-1;
  const float* Arow = q + bidx * qBS + (size_t)am * H_;
  const unsigned short* Brow = rows + bidx * rBS + (size_t)(r0 + mrow) * H_;
  v8f acc = {0.f,0.f,0.f,0.f,0.f,0.f,0.f,0.f};
  for (int k0 = 0; k0 < H_; k0 += 32){
    ABreg a, b;
    int kb = k0 + half * 16;
    #pragma unroll
    for (int j = 0; j < 8; ++j){
      int ka = k0 + 2*j + half * 16;
      a.u[j] = bfpair(Arow[ka], Arow[ka+1]);
      b.u[j] = *(const unsigned int*)(Brow + kb + 2*j);
    }
    acc = __builtin_amdgcn_wmma_f32_16x16x32_bf16(false, a.v, false, b.v, (short)0, acc, false, false);
  }
  #pragma unroll
  for (int j = 0; j < 8; ++j){
    int m = j + half * 8;
    int n = r0 + mrow;
    if (m < R) out[bidx * outBS + (size_t)m * Ncols + n] = acc[j] * scale;
  }
}

// ---------------- softmax over LP, write bf16 weights ----------------
__global__ void k_softmax(const float* __restrict__ in, unsigned short* __restrict__ out,
                          long bstride, int rstride){
  __shared__ float red[256];
  int tid = threadIdx.x;
  long base = blockIdx.y * bstride + (long)blockIdx.x * rstride;
  float v[8];
  float mx = -3.4e38f;
  #pragma unroll
  for (int i = 0; i < 8; ++i){ v[i] = in[base + tid + i*256]; mx = fmaxf(mx, v[i]); }
  red[tid] = mx; __syncthreads();
  for (int o = 128; o > 0; o >>= 1){ if (tid < o) red[tid] = fmaxf(red[tid], red[tid+o]); __syncthreads(); }
  mx = red[0]; __syncthreads();
  float s = 0.f;
  #pragma unroll
  for (int i = 0; i < 8; ++i){ v[i] = __expf(v[i] - mx); s += v[i]; }
  red[tid] = s; __syncthreads();
  for (int o = 128; o > 0; o >>= 1){ if (tid < o) red[tid] += red[tid+o]; __syncthreads(); }
  float inv = 1.0f / red[0];
  #pragma unroll
  for (int i = 0; i < 8; ++i) out[base + tid + i*256] = f2bf(v[i] * inv);
}

// ---------------- ctx: out[b][m][h] = sum_l w[b][m][l]*pv[b][l][h]; K=LP_, N=H_ ----------------
// pv tile staged with CDNA5 async LDS DMA (global_load_async_to_lds_b128, ASYNCcnt).
__global__ __launch_bounds__(256) void k_attn_ctx(
    const unsigned short* __restrict__ w, long wBS, int R,
    const unsigned short* __restrict__ pv, long pvBS,
    float* __restrict__ out, long outBS){
  __shared__ unsigned short ldsB[32 * 128];      // raw row-major [k][n] bf16 tile (8KB)
  int tid = threadIdx.x;
  int lane = tid & 31, wave = tid >> 5;
  int half = lane >> 4, mrow = lane & 15;
  int b = blockIdx.y;
  int nblk = blockIdx.x * 128;
  int nsub = wave * 16;
  int am = mrow; if (am > R-1) am = R-1;
  const unsigned short* Arow = w + b * wBS + (size_t)am * LP_;
  unsigned int ldsBase = (unsigned int)(size_t)(void*)ldsB;  // LDS byte offset (flat addr low 32b)
  v8f acc = {0.f,0.f,0.f,0.f,0.f,0.f,0.f,0.f};
  for (int k0 = 0; k0 < LP_; k0 += 32){
    // async DMA: 8KB tile = 512 x 16B; each of 256 threads issues two b128 copies
    #pragma unroll
    for (int i = 0; i < 2; ++i){
      int e = tid + i * 256;                     // 0..511
      int kk = e >> 4;                           // tile row 0..31
      int cu = (e & 15) * 8;                     // u16 column within row (steps of 8 = 16B)
      const unsigned short* gsrc = pv + b * pvBS + (size_t)(k0 + kk) * H_ + nblk + cu;
      unsigned int ldsaddr = ldsBase + (unsigned int)e * 16u;
      unsigned long long ga = (unsigned long long)(size_t)gsrc;
      asm volatile("global_load_async_to_lds_b128 %0, %1, off"
                   :: "v"(ldsaddr), "v"(ga) : "memory");
    }
    asm volatile("s_wait_asynccnt 0x0" ::: "memory");
    __syncthreads();
    ABreg a, bb;
    #pragma unroll
    for (int j = 0; j < 8; ++j){
      int ka = k0 + 2*j + half * 16;
      a.u[j]  = *(const unsigned int*)(Arow + ka);
      int kb = half * 16 + 2*j;                  // B k-pair within tile
      unsigned int lo = ldsB[kb * 128 + nsub + mrow];
      unsigned int hi = ldsB[(kb + 1) * 128 + nsub + mrow];
      bb.u[j] = lo | (hi << 16);
    }
    acc = __builtin_amdgcn_wmma_f32_16x16x32_bf16(false, a.v, false, bb.v, (short)0, acc, false, false);
    __syncthreads();
  }
  #pragma unroll
  for (int j = 0; j < 8; ++j){
    int m = j + half * 8;
    int n = nblk + nsub + mrow;
    if (m < R) out[b * outBS + (size_t)m * H_ + n] = acc[j];
  }
}

// ---------------- argmin over codebook: key[c] = (-2*prod)[b][c] + cb_sq[c] ----------------
__global__ void k_argmin(const float* __restrict__ prodm2, const float* __restrict__ cbsq,
                         int* __restrict__ opidx){
  __shared__ float sv[256]; __shared__ int si[256];
  int b = blockIdx.x, tid = threadIdx.x;
  float bv = 3.4e38f; int bi = 0;
  for (int c = tid; c < C_; c += 256){
    float v = prodm2[(size_t)b * C_ + c] + cbsq[c];
    if (v < bv){ bv = v; bi = c; }
  }
  sv[tid] = bv; si[tid] = bi; __syncthreads();
  for (int o = 128; o > 0; o >>= 1){
    if (tid < o){
      if (sv[tid+o] < sv[tid] || (sv[tid+o] == sv[tid] && si[tid+o] < si[tid])){
        sv[tid] = sv[tid+o]; si[tid] = si[tid+o];
      }
    }
    __syncthreads();
  }
  if (tid == 0) opidx[b] = si[0];
}

// ---------------- slot seeds ----------------
__global__ void k_slotseed(const float* __restrict__ sctx, const float* __restrict__ slotq,
                           float* __restrict__ seed){
  int idx = blockIdx.x * 256 + threadIdx.x;       // < B_*S1_*H_
  int h = idx & (H_ - 1);
  int s = (idx >> 10) % S1_;
  int b = idx / (S1_ * H_);
  seed[idx] = sctx[b * H_ + h] + slotq[s * H_ + h];
}

// ---------------- gating matvec + sigmoid ----------------
__global__ void k_gate(const float* __restrict__ seed, const float* __restrict__ Wg,
                       const float* __restrict__ bg, float* __restrict__ probs){
  int tid = threadIdx.x, lane = tid & 31, wr = tid >> 5;
  int row = blockIdx.x * 8 + wr;                  // < 144
  const float* r = seed + (size_t)row * H_;
  float s = 0.f;
  for (int h = lane; h < H_; h += 32) s += r[h] * Wg[h];
  for (int o = 16; o > 0; o >>= 1) s += __shfl_xor(s, o, 32);
  if (lane == 0) probs[row] = sigmoidf_(s + bg[0]);
}

// ---------------- mask + global any() fallback ----------------
__global__ void k_mask(const float* __restrict__ probs, float* __restrict__ mask,
                       float* __restrict__ cnt){
  __shared__ float p[B_ * S1_];
  __shared__ int anyv;
  int tid = threadIdx.x;
  if (tid < B_ * S1_) p[tid] = probs[tid];
  __syncthreads();
  if (tid == 0){
    int a = 0;
    for (int i = 0; i < B_ * S1_; ++i) if (p[i] >= 0.5f) a = 1;
    anyv = a;
  }
  __syncthreads();
  if (tid < B_){
    int b = tid;
    float s = 0.f; int arg = 0; float bv = p[b * S1_];
    for (int sl = 0; sl < S1_; ++sl){
      float v = p[b * S1_ + sl];
      if (v > bv){ bv = v; arg = sl; }
      if (v >= 0.5f) s += 1.f;
    }
    for (int sl = 0; sl < S1_; ++sl){
      float m = anyv ? ((p[b * S1_ + sl] >= 0.5f) ? 1.f : 0.f)
                     : ((sl == arg) ? 1.f : 0.f);
      mask[b * S1_ + sl] = m;
    }
    cnt[b] = anyv ? fmaxf(s, 1.f) : 1.f;
  }
}

// ---------------- matrix summary = tanh(cb[opidx] + masked-mean(slot_ctx)) ----------------
__global__ void k_msum(const float* __restrict__ mask, const float* __restrict__ cnt,
                       const float* __restrict__ sctx9, const float* __restrict__ cb,
                       const int* __restrict__ opidx, float* __restrict__ msum,
                       float* __restrict__ dout, int t){
  int h = blockIdx.x * 256 + threadIdx.x;
  int b = blockIdx.y;
  float s = 0.f;
  #pragma unroll
  for (int sl = 0; sl < S1_; ++sl)
    s += mask[b * S1_ + sl] * sctx9[((size_t)b * S1_ + sl) * H_ + h];
  s /= cnt[b];
  float v = tanhf(cb[(size_t)opidx[b] * H_ + h] + s);
  msum[b * H_ + h] = v;
  dout[((size_t)b * T_ + t) * H_ + h] = v;
}

// ---------------- stop logit / prob ----------------
__global__ void k_stop(const float* __restrict__ sctx, const float* __restrict__ msum,
                       const float* __restrict__ Ws, const float* __restrict__ bs,
                       float* __restrict__ dout, int t){
  int tid = threadIdx.x, lane = tid & 31, wr = tid >> 5;
  int b = blockIdx.x * 8 + wr;                    // 2 blocks x 8 waves = 16
  float s = 0.f;
  for (int h = lane; h < H_; h += 32)
    s += sctx[b * H_ + h] * Ws[h] + msum[b * H_ + h] * Ws[H_ + h];
  for (int o = 16; o > 0; o >>= 1) s += __shfl_xor(s, o, 32);
  if (lane == 0){
    float lg = s + bs[0];
    dout[OFF_LOG + b * T_ + t] = lg;
    dout[OFF_PRB + b * T_ + t] = sigmoidf_(lg);
  }
}

// ---------------- GRU pointwise combine (in-place state update) ----------------
__global__ void k_gru(const float* __restrict__ gi, const float* __restrict__ gh,
                      float* __restrict__ state){
  int idx = blockIdx.x * 256 + threadIdx.x;       // < B_*H_
  int b = idx >> 10, h = idx & (H_ - 1);
  const float* gib = gi + (size_t)b * 3 * H_;
  const float* ghb = gh + (size_t)b * 3 * H_;
  float r = sigmoidf_(gib[h] + ghb[h]);
  float z = sigmoidf_(gib[H_ + h] + ghb[H_ + h]);
  float n = tanhf(gib[2*H_ + h] + r * ghb[2*H_ + h]);
  state[idx] = (1.f - z) * n + z * state[idx];
}

// ---------------- chain lengths ----------------
__global__ void k_chain(float* __restrict__ dout){
  int b = threadIdx.x;
  if (b >= B_) return;
  int len = T_;
  for (int t = 0; t < T_; ++t){
    if (dout[OFF_PRB + b * T_ + t] >= 0.5f){ len = t + 1; break; }
  }
  dout[OFF_CHN + b] = (float)len;
}

extern "C" void kernel_launch(void* const* d_in, const int* in_sizes, int n_in,
                              void* d_out, int out_size, void* d_ws, size_t ws_size,
                              hipStream_t stream){
  const float* logic  = (const float*)d_in[0];
  const float* prompt = (const float*)d_in[1];
  const float* cb     = (const float*)d_in[2];
  const float* Winit  = (const float*)d_in[3];
  const float* Wpq    = (const float*)d_in[4];
  const float* Wpk    = (const float*)d_in[5];
  const float* Wpv    = (const float*)d_in[6];
  const float* slotq  = (const float*)d_in[7];
  const float* Wsq    = (const float*)d_in[8];
  const float* Wop    = (const float*)d_in[9];
  const float* Wg     = (const float*)d_in[10];
  const float* bg     = (const float*)d_in[11];
  const float* Wst    = (const float*)d_in[12];
  const float* bst    = (const float*)d_in[13];
  const float* Wih    = (const float*)d_in[14];
  const float* Whh    = (const float*)d_in[15];
  const float* bih    = (const float*)d_in[16];
  const float* bhh    = (const float*)d_in[17];
  float* dout = (float*)d_out;

  // ---- workspace carve (~160 MB; deterministic, recomputed every call) ----
  char* p = (char*)d_ws;
  auto carve = [&](size_t bytes)->char*{
    char* r = p; p += (bytes + 255) & ~((size_t)255); return r; };
  float* cat    = (float*)carve(sizeof(float) * B_ * 2 * H_);
  float* state  = (float*)carve(sizeof(float) * B_ * H_);
  float* stateq = (float*)carve(sizeof(float) * B_ * H_);
  float* sctx   = (float*)carve(sizeof(float) * B_ * H_);
  float* oppre  = (float*)carve(sizeof(float) * B_ * H_);
  float* prod   = (float*)carve(sizeof(float) * B_ * C_);
  float* cbsq   = (float*)carve(sizeof(float) * C_);
  float* sseed  = (float*)carve(sizeof(float) * B_ * S1_ * H_);
  float* sq     = (float*)carve(sizeof(float) * B_ * S1_ * H_);
  float* sctx9  = (float*)carve(sizeof(float) * B_ * S1_ * H_);
  float* scores = (float*)carve(sizeof(float) * B_ * S1_ * LP_);
  float* maskb  = (float*)carve(sizeof(float) * B_ * S1_);
  float* cnt    = (float*)carve(sizeof(float) * B_);
  float* probs  = (float*)carve(sizeof(float) * B_ * S1_);
  float* msum   = (float*)carve(sizeof(float) * B_ * H_);
  float* gi     = (float*)carve(sizeof(float) * B_ * 3 * H_);
  float* gh     = (float*)carve(sizeof(float) * B_ * 3 * H_);
  int*   opidx  = (int*)  carve(sizeof(int) * B_);
  unsigned short* pk_bf = (unsigned short*)carve((size_t)2 * B_ * LP_ * H_);
  unsigned short* pv_bf = (unsigned short*)carve((size_t)2 * B_ * LP_ * H_);
  unsigned short* cb_bf = (unsigned short*)carve((size_t)2 * C_ * H_);
  unsigned short* w_bf  = (unsigned short*)carve((size_t)2 * B_ * S1_ * LP_);

  dim3 blk(256);
  const float scl = 0.03125f;                  // 1/sqrt(H)
  const long  LPH = (long)LP_ * H_;
  const long  SLP = (long)S1_ * LP_;

  // ---- one-time precompute ----
  k_mean  <<<dim3(H_/256, B_), blk, 0, stream>>>(prompt, LP_, cat, 2*H_, 0);
  k_mean  <<<dim3(H_/256, B_), blk, 0, stream>>>(logic,  LL_, cat, 2*H_, H_);
  k_cbprep<<<dim3(C_),         blk, 0, stream>>>(cb, cb_bf, cbsq);
  k_gemm_small<<<dim3(8, 1),   blk, 0, stream>>>(cat, B_, 2*H_, Winit, H_, nullptr, state, 1);
  k_proj  <<<dim3(8, 512),     blk, 0, stream>>>(prompt, Wpk, pk_bf);
  k_proj  <<<dim3(8, 512),     blk, 0, stream>>>(prompt, Wpv, pv_bf);

  // ---- T-step autoregressive chain ----
  for (int t = 0; t < T_; ++t){
    // state attention
    k_gemm_small <<<dim3(8, 1),        blk, 0, stream>>>(state, B_, H_, Wpq, H_, nullptr, stateq, 0);
    k_attn_scores<<<dim3(LP_/128, B_), blk, 0, stream>>>(stateq, 1, (long)H_, pk_bf, LPH, LP_, scores, SLP, scl);
    k_softmax    <<<dim3(1, B_),       blk, 0, stream>>>(scores, w_bf, SLP, LP_);
    k_attn_ctx   <<<dim3(8, B_),       blk, 0, stream>>>(w_bf, SLP, 1, pv_bf, LPH, sctx, (long)H_);
    // codebook op selection
    k_gemm_small <<<dim3(8, 1),        blk, 0, stream>>>(sctx, B_, H_, Wop, H_, nullptr, oppre, 0);
    k_attn_scores<<<dim3(C_/128, 1),   blk, 0, stream>>>(oppre, 16, 0L, cb_bf, 0L, C_, prod, 0L, -2.0f);
    k_argmin     <<<dim3(B_),          blk, 0, stream>>>(prod, cbsq, opidx);
    // slot attention
    k_slotseed   <<<dim3(B_*S1_*H_/256), blk, 0, stream>>>(sctx, slotq, sseed);
    k_gemm_small <<<dim3(8, S1_),      blk, 0, stream>>>(sseed, B_*S1_, H_, Wsq, H_, nullptr, sq, 0);
    k_attn_scores<<<dim3(LP_/128, B_), blk, 0, stream>>>(sq, S1_, (long)S1_*H_, pk_bf, LPH, LP_, scores, SLP, scl);
    k_softmax    <<<dim3(S1_, B_),     blk, 0, stream>>>(scores, w_bf, SLP, LP_);
    k_attn_ctx   <<<dim3(8, B_),       blk, 0, stream>>>(w_bf, SLP, S1_, pv_bf, LPH, sctx9, (long)S1_*H_);
    // gating + summary + stop
    k_gate       <<<dim3(B_*S1_/8),    blk, 0, stream>>>(sseed, Wg, bg, probs);
    k_mask       <<<dim3(1),           blk, 0, stream>>>(probs, maskb, cnt);
    k_msum       <<<dim3(H_/256, B_),  blk, 0, stream>>>(maskb, cnt, sctx9, cb, opidx, msum, dout, t);
    k_stop       <<<dim3(2),           blk, 0, stream>>>(sctx, msum, Wst, bst, dout, t);
    // GRU update
    k_gemm_small <<<dim3(3*H_/128, 1), blk, 0, stream>>>(msum,  B_, H_, Wih, 3*H_, bih, gi, 0);
    k_gemm_small <<<dim3(3*H_/128, 1), blk, 0, stream>>>(state, B_, H_, Whh, 3*H_, bhh, gh, 0);
    k_gru        <<<dim3(B_*H_/256),   blk, 0, stream>>>(gi, gh, state);
  }
  k_chain<<<dim3(1), dim3(32), 0, stream>>>(dout);
}
